// TabularDiffFlow_86028194939128
// MI455X (gfx1250) — matrix operand order repaired and checked by reading, not compiled, not run on hardware
//
#include <hip/hip_runtime.h>
#include <hip/hip_bf16.h>
#include <math.h>

// ---------------------------------------------------------------------------
// Types for CDNA5 WMMA (wave32): D(f32 16x16) = A(bf16 16x32) x B(bf16 32x16) + C
// ---------------------------------------------------------------------------
typedef __attribute__((ext_vector_type(16))) __bf16 v16bf;
typedef __attribute__((ext_vector_type(8)))  float  v8f;

union BFrag { uint4 u[2]; v16bf v; };

__device__ __forceinline__ unsigned short f2bf(float x) {
  unsigned u = __float_as_uint(x);
  unsigned r = u + 0x7FFFu + ((u >> 16) & 1u);   // round-to-nearest-even
  return (unsigned short)(r >> 16);
}
__device__ __forceinline__ unsigned pack2(float a, float b) {
  return (unsigned)f2bf(a) | ((unsigned)f2bf(b) << 16);
}
__device__ __forceinline__ v8f wmma_bf16(v16bf a, v16bf b, v8f c) {
  return __builtin_amdgcn_wmma_f32_16x16x32_bf16(false, a, false, b, (short)0, c,
                                                 false, false);
}
// A fragment: ISA 16-bit A 16x32 layout. lane<16: row=lane&15, halves {0..7,16..23};
// lane>=16: halves {8..15,24..31}. base indexed in uint4 (8 halves).
__device__ __forceinline__ v16bf load_a_frag(const uint4* base, int stride_u4,
                                             int row, int k_u4, int lane) {
  int o = lane >> 4;
  BFrag f;
  f.u[0] = base[row * stride_u4 + k_u4 + o];
  f.u[1] = base[row * stride_u4 + k_u4 + 2 + o];
  return f.v;
}
// B fragment: 32x16, column per lane; lane<16 holds k{0..7,16..23}, lane>=16 holds k{8..15,24..31}.
// LDS holds W transposed as [n][k], so this is two contiguous uint4 per lane.
__device__ __forceinline__ v16bf load_b_frag(const uint4* base, int stride_u4,
                                             int col, int k_u4, int lane) {
  int o = lane >> 4;
  BFrag f;
  f.u[0] = base[col * stride_u4 + k_u4 + o];
  f.u[1] = base[col * stride_u4 + k_u4 + 2 + o];
  return f.v;
}
__device__ __forceinline__ float silu_f(float v) { return v / (1.f + expf(-v)); }
__device__ __forceinline__ unsigned fkey(float f) {  // order-preserving float->uint
  unsigned u = __float_as_uint(f);
  return (u >> 31) ? ~u : (u | 0x80000000u);
}

// ---------------------------------------------------------------------------
// Generic GEMM: C[M,N] = act(A[M,K] @ W[K,N] + bias), fp32 I/O, bf16 WMMA math.
// M,N multiples of 128; K multiple of 32. Block tile 128x128, 8 waves (2x4),
// wave tile 64x32 -> 8 WMMA accumulators per wave.
// ---------------------------------------------------------------------------
template <int ACT>
__global__ __launch_bounds__(256) void gemm_kernel(const float* __restrict__ A,
                                                   const float* __restrict__ W,
                                                   const float* __restrict__ bias,
                                                   float* __restrict__ C,
                                                   int M, int N, int K) {
  __shared__ __align__(16) unsigned short As[128 * 32];  // [m][k] bf16
  __shared__ __align__(16) unsigned short Ws[128 * 32];  // [n][k] bf16 (transposed)
  const int tid = threadIdx.x, lane = tid & 31, w = tid >> 5;
  const int wm = w >> 2, wn = w & 3;
  const int n0 = blockIdx.x * 128, m0 = blockIdx.y * 128;
  (void)M;
  v8f acc[4][2] = {};
  uint4* Asu = (uint4*)As;
  uint4* Wsu = (uint4*)Ws;

  for (int kk = 0; kk < K; kk += 32) {
    {  // stage A tile 128x32, convert fp32 -> bf16
      int r = tid >> 1, half = tid & 1;
      const float4* ga = (const float4*)(A + (size_t)(m0 + r) * K + kk + half * 16);
      __builtin_prefetch(A + (size_t)(m0 + r) * K + kk + 32, 0, 1);
      float4 f0 = ga[0], f1 = ga[1], f2 = ga[2], f3 = ga[3];
      uint4 p0, p1;
      p0.x = pack2(f0.x, f0.y); p0.y = pack2(f0.z, f0.w);
      p0.z = pack2(f1.x, f1.y); p0.w = pack2(f1.z, f1.w);
      p1.x = pack2(f2.x, f2.y); p1.y = pack2(f2.z, f2.w);
      p1.z = pack2(f3.x, f3.y); p1.w = pack2(f3.z, f3.w);
      Asu[r * 4 + half * 2] = p0;
      Asu[r * 4 + half * 2 + 1] = p1;
    }
    {  // stage W tile 32x128 transposed into [n][k]; loads coalesced across lanes
      int n = tid & 127, ks = tid >> 7;  // ks in {0,1} -> k offset 16*ks
      const float* gw = W + (size_t)(kk + ks * 16) * N + n0 + n;
      unsigned u[8];
#pragma unroll
      for (int j = 0; j < 8; ++j) {
        float a = gw[(size_t)(2 * j) * N];
        float b = gw[(size_t)(2 * j + 1) * N];
        u[j] = pack2(a, b);
      }
      uint4 q0, q1;
      q0.x = u[0]; q0.y = u[1]; q0.z = u[2]; q0.w = u[3];
      q1.x = u[4]; q1.y = u[5]; q1.z = u[6]; q1.w = u[7];
      Wsu[n * 4 + ks * 2] = q0;
      Wsu[n * 4 + ks * 2 + 1] = q1;
    }
    __syncthreads();
    v16bf b0 = load_b_frag(Wsu, 4, wn * 32 + (lane & 15), 0, lane);
    v16bf b1 = load_b_frag(Wsu, 4, wn * 32 + 16 + (lane & 15), 0, lane);
#pragma unroll
    for (int mi = 0; mi < 4; ++mi) {
      v16bf a = load_a_frag(Asu, 4, wm * 64 + mi * 16 + (lane & 15), 0, lane);
      acc[mi][0] = wmma_bf16(a, b0, acc[mi][0]);
      acc[mi][1] = wmma_bf16(a, b1, acc[mi][1]);
    }
    __syncthreads();
  }
  // epilogue: C/D layout -> m = tile + r + 8*(lane>=16), n = tile + (lane&15)
  const int hi = (lane >> 4) * 8;
#pragma unroll
  for (int mi = 0; mi < 4; ++mi)
#pragma unroll
    for (int ni = 0; ni < 2; ++ni) {
      int col = n0 + wn * 32 + ni * 16 + (lane & 15);
      float bvv = bias ? bias[col] : 0.f;
#pragma unroll
      for (int r = 0; r < 8; ++r) {
        int m = m0 + wm * 64 + mi * 16 + r + hi;
        float v = acc[mi][ni][r] + bvv;
        if (ACT == 1) v = silu_f(v);
        C[(size_t)m * N + col] = v;
      }
    }
}

// ---------------------------------------------------------------------------
// Attention per (b,h): builds Q/K/Vt (bf16) from rank-structured q = qh[b]+qf[f],
// S = Q K^T / 8 via WMMA, exact top-64 threshold per row (radix select), softmax,
// O = P V via WMMA, emits column mean into pooled[b, h*64 + d].
// LDS: Q 16K | K 16K | Vt 16K | S 64K = 112K (of 320K/WGP). P reuses Q+K region.
// ---------------------------------------------------------------------------
__global__ __launch_bounds__(256) void attn_kernel(const float* __restrict__ qh,
                                                   const float* __restrict__ qf,
                                                   const float* __restrict__ kh,
                                                   const float* __restrict__ kf,
                                                   const float* __restrict__ vh,
                                                   const float* __restrict__ vf,
                                                   float* __restrict__ pooled) {
  __shared__ __align__(16) unsigned char smem[114688];
  unsigned short* Qs = (unsigned short*)smem;             // [128][64] bf16
  unsigned short* Ks = (unsigned short*)(smem + 16384);   // [128][64] bf16
  unsigned short* Vt = (unsigned short*)(smem + 32768);   // [64][128] bf16 (V^T)
  float* S = (float*)(smem + 49152);                      // [128][128] f32
  unsigned short* Pb = (unsigned short*)smem;             // [128][128] bf16 (reuse)
  float* Of = (float*)(smem + 49152);                     // [128][64] f32 (reuse S)

  const int h = blockIdx.x, b = blockIdx.y;
  const int tid = threadIdx.x, lane = tid & 31, w = tid >> 5;
  const int cb = h * 64;

  // stage Q and K: q[b,f,d] = qh[b,d'] + qf[f,d']
  for (int i = tid; i < 128 * 32; i += 256) {
    int f = i >> 5, d = (i & 31) * 2;
    float q0 = qh[b * 512 + cb + d] + qf[f * 512 + cb + d];
    float q1 = qh[b * 512 + cb + d + 1] + qf[f * 512 + cb + d + 1];
    ((unsigned*)Qs)[f * 32 + (d >> 1)] = pack2(q0, q1);
    float k0 = kh[b * 512 + cb + d] + kf[f * 512 + cb + d];
    float k1 = kh[b * 512 + cb + d + 1] + kf[f * 512 + cb + d + 1];
    ((unsigned*)Ks)[f * 32 + (d >> 1)] = pack2(k0, k1);
  }
  // stage V transposed: Vt[d][key]
  for (int i = tid; i < 64 * 64; i += 256) {
    int d = i & 63, kp = i >> 6, key = kp * 2;
    float vhd = vh[b * 512 + cb + d];
    float v0 = vhd + vf[(size_t)key * 512 + cb + d];
    float v1 = vhd + vf[(size_t)(key + 1) * 512 + cb + d];
    ((unsigned*)Vt)[d * 64 + kp] = pack2(v0, v1);
  }
  __syncthreads();

  // S = (Q K^T) * 1/sqrt(64); wave w owns query rows [w*16, w*16+16)
  {
    v8f acc[8] = {};
    const uint4* Qu = (const uint4*)Qs;  // stride 8 uint4 per row
    const uint4* Ku = (const uint4*)Ks;
#pragma unroll
    for (int ks = 0; ks < 2; ++ks) {
      v16bf a = load_a_frag(Qu, 8, w * 16 + (lane & 15), ks * 4, lane);
#pragma unroll
      for (int ni = 0; ni < 8; ++ni) {
        v16bf bb = load_b_frag(Ku, 8, ni * 16 + (lane & 15), ks * 4, lane);
        acc[ni] = wmma_bf16(a, bb, acc[ni]);
      }
    }
    const int hi = (lane >> 4) * 8;
#pragma unroll
    for (int ni = 0; ni < 8; ++ni)
#pragma unroll
      for (int r = 0; r < 8; ++r)
        S[(w * 16 + r + hi) * 128 + ni * 16 + (lane & 15)] = acc[ni][r] * 0.125f;
  }
  __syncthreads();

  // per-row exact 64th-largest threshold (radix select in ordered-uint space),
  // mask (score < thr -> dropped), softmax; write P as bf16 over dead Q/K LDS
  for (int row = w * 16; row < w * 16 + 16; ++row) {
    float s[4];
    unsigned kk[4];
#pragma unroll
    for (int q = 0; q < 4; ++q) {
      s[q] = S[row * 128 + lane + 32 * q];
      kk[q] = fkey(s[q]);
    }
    unsigned prefix = 0;
    for (int bit = 31; bit >= 0; --bit) {
      unsigned cand = prefix | (1u << bit);
      int c = 0;
#pragma unroll
      for (int q = 0; q < 4; ++q) c += (kk[q] >= cand) ? 1 : 0;
      for (int off = 16; off > 0; off >>= 1) c += __shfl_xor(c, off, 32);
      if (c >= 64) prefix = cand;  // max v with count(>=v) >= 64 == 64th largest
    }
    float m = fmaxf(fmaxf(s[0], s[1]), fmaxf(s[2], s[3]));
    for (int off = 16; off > 0; off >>= 1) m = fmaxf(m, __shfl_xor(m, off, 32));
    float p[4], sum = 0.f;
#pragma unroll
    for (int q = 0; q < 4; ++q) {
      p[q] = (kk[q] >= prefix) ? expf(s[q] - m) : 0.f;  // masked -> exp(-1e9-m)=0
      sum += p[q];
    }
    for (int off = 16; off > 0; off >>= 1) sum += __shfl_xor(sum, off, 32);
    float inv = 1.f / sum;
#pragma unroll
    for (int q = 0; q < 4; ++q)
      Pb[row * 128 + lane + 32 * q] = f2bf(p[q] * inv);
  }
  __syncthreads();

  // O = P @ V  (A = P [128x128], B = V via Vt [d][key]); then column mean
  {
    v8f acc[4] = {};
    const uint4* Pu = (const uint4*)Pb;  // stride 16 uint4 per row
    const uint4* Vu = (const uint4*)Vt;  // stride 16 uint4 per d-row
#pragma unroll
    for (int ks = 0; ks < 4; ++ks) {
      v16bf a = load_a_frag(Pu, 16, w * 16 + (lane & 15), ks * 4, lane);
#pragma unroll
      for (int ni = 0; ni < 4; ++ni) {
        v16bf bb = load_b_frag(Vu, 16, ni * 16 + (lane & 15), ks * 4, lane);
        acc[ni] = wmma_bf16(a, bb, acc[ni]);
      }
    }
    const int hi = (lane >> 4) * 8;
#pragma unroll
    for (int ni = 0; ni < 4; ++ni)
#pragma unroll
      for (int r = 0; r < 8; ++r)
        Of[(w * 16 + r + hi) * 64 + ni * 16 + (lane & 15)] = acc[ni][r];
  }
  __syncthreads();
  if (tid < 64) {
    float sum = 0.f;
    for (int f = 0; f < 128; ++f) sum += Of[f * 64 + tid];
    pooled[b * 512 + cb + tid] = sum * (1.f / 128.f);
  }
}

// ---------------------------------------------------------------------------
// Small elementwise / norm kernels
// ---------------------------------------------------------------------------
__global__ void time_embed_hidden(const float* __restrict__ t,
                                  const float* __restrict__ w1,
                                  const float* __restrict__ b1,
                                  float* __restrict__ out) {
  int i = blockIdx.x * 256 + threadIdx.x;  // 512*512
  int b = i >> 9, d = i & 511;
  out[i] = silu_f(t[b] * w1[d] + b1[d]);
}
__global__ void concat2(const float* __restrict__ a, const float* __restrict__ b,
                        float* __restrict__ c) {
  int i = blockIdx.x * 256 + threadIdx.x;  // 512*512
  int r = i >> 9, col = i & 511;
  c[r * 1024 + col] = a[i];
  c[r * 1024 + 512 + col] = b[i];
}
__global__ void make_fs(const float* __restrict__ fid, float* __restrict__ fs) {
  int i = blockIdx.x * 256 + threadIdx.x;  // 128*512
  int f = i >> 9, d = i & 511;
  fs[i] = (d < 128) ? 0.1f * fid[f * 128 + d] : 0.f;
}
__global__ __launch_bounds__(256) void ln_kernel(const float* __restrict__ x,
                                                 const float* __restrict__ g,
                                                 const float* __restrict__ bta,
                                                 float* __restrict__ y) {
  __shared__ float red[256];
  int b = blockIdx.x, t = threadIdx.x;
  float a = x[b * 512 + t], c = x[b * 512 + 256 + t];
  red[t] = a + c;
  __syncthreads();
  for (int s = 128; s > 0; s >>= 1) {
    if (t < s) red[t] += red[t + s];
    __syncthreads();
  }
  float mu = red[0] * (1.f / 512.f);
  __syncthreads();
  float da = a - mu, dc = c - mu;
  red[t] = da * da + dc * dc;
  __syncthreads();
  for (int s = 128; s > 0; s >>= 1) {
    if (t < s) red[t] += red[t + s];
    __syncthreads();
  }
  float inv = rsqrtf(red[0] * (1.f / 512.f) + 1e-5f);
  y[b * 512 + t] = da * inv * g[t] + bta[t];
  y[b * 512 + 256 + t] = dc * inv * g[256 + t] + bta[256 + t];
}
__global__ void tanh_head(const float* __restrict__ r3, float* __restrict__ out) {
  int i = blockIdx.x * 256 + threadIdx.x;  // 512*128
  int b = i >> 7, f = i & 127;
  out[i] = tanhf(r3[b * 256 + f]);
}

// ---------------------------------------------------------------------------
// Launcher
// ---------------------------------------------------------------------------
extern "C" void kernel_launch(void* const* d_in, const int* in_sizes, int n_in,
                              void* d_out, int out_size, void* d_ws, size_t ws_size,
                              hipStream_t stream) {
  (void)in_sizes; (void)n_in; (void)out_size; (void)ws_size;
  const float* x_t   = (const float*)d_in[0];
  const float* t     = (const float*)d_in[1];
  const float* te_w1 = (const float*)d_in[2];
  const float* te_b1 = (const float*)d_in[3];
  const float* te_w2 = (const float*)d_in[4];
  const float* te_b2 = (const float*)d_in[5];
  const float* fe_w1 = (const float*)d_in[6];
  const float* fe_b1 = (const float*)d_in[7];
  const float* fe_w2 = (const float*)d_in[8];
  const float* fe_b2 = (const float*)d_in[9];
  const float* fp_w1 = (const float*)d_in[10];
  const float* fp_b1 = (const float*)d_in[11];
  const float* fp_w2 = (const float*)d_in[12];
  const float* fp_b2 = (const float*)d_in[13];
  const float* fp_w3 = (const float*)d_in[14];
  const float* fp_b3 = (const float*)d_in[15];
  const float* fid   = (const float*)d_in[16];
  const float* wq = (const float*)d_in[17];
  const float* bq = (const float*)d_in[18];
  const float* wk = (const float*)d_in[19];
  const float* bk = (const float*)d_in[20];
  const float* wv = (const float*)d_in[21];
  const float* bvp = (const float*)d_in[22];
  const float* wo = (const float*)d_in[23];
  const float* bo = (const float*)d_in[24];
  const float* gp_w1 = (const float*)d_in[25];
  const float* gp_b1 = (const float*)d_in[26];
  const float* gp_w2 = (const float*)d_in[27];
  const float* gp_b2 = (const float*)d_in[28];
  const float* ln_g  = (const float*)d_in[29];
  const float* ln_b  = (const float*)d_in[30];
  const float* dn_w1 = (const float*)d_in[31];
  const float* dn_b1 = (const float*)d_in[32];
  const float* dn_w2 = (const float*)d_in[33];
  const float* dn_b2 = (const float*)d_in[34];
  const float* dn_w3 = (const float*)d_in[35];
  const float* dn_b3 = (const float*)d_in[36];

  float* p = (float*)d_ws;
  auto alloc = [&](size_t n) { float* r = p; p += n; return r; };
  const size_t BD = 512 * 512;
  float* x_hid  = alloc(BD);
  float* x_emb  = alloc(BD);
  float* t_hid  = alloc(BD);
  float* t_emb  = alloc(BD);
  float* comb   = alloc(512 * 1024);
  float* h1     = alloc(BD);
  float* h2     = alloc(BD);
  float* h3     = alloc(BD);
  float* fs     = alloc(128 * 512);
  float* qh     = alloc(BD);
  float* khb    = alloc(BD);
  float* vhb    = alloc(BD);
  float* qf     = alloc(128 * 512);
  float* kfb    = alloc(128 * 512);
  float* vfb    = alloc(128 * 512);
  float* pooled = alloc(BD);
  float* g0     = alloc(BD);
  float* gp1    = alloc(512 * 1024);
  float* g2     = alloc(BD);
  float* gn     = alloc(BD);
  float* r1     = alloc(512 * 1024);
  float* r2     = alloc(512 * 1024);
  float* r3     = alloc(512 * 256);

  dim3 blk(256);
  auto gemm = [&](const float* A, const float* W, const float* bias, float* C,
                  int M, int N, int K, int act) {
    dim3 g(N / 128, M / 128);
    if (act)
      gemm_kernel<1><<<g, blk, 0, stream>>>(A, W, bias, C, M, N, K);
    else
      gemm_kernel<0><<<g, blk, 0, stream>>>(A, W, bias, C, M, N, K);
  };

  // embeddings
  time_embed_hidden<<<1024, blk, 0, stream>>>(t, te_w1, te_b1, t_hid);
  gemm(x_t, fe_w1, fe_b1, x_hid, 512, 512, 128, 1);    // silu(x@fe_w1+b)
  gemm(x_hid, fe_w2, fe_b2, x_emb, 512, 512, 512, 0);
  gemm(t_hid, te_w2, te_b2, t_emb, 512, 512, 512, 0);
  concat2<<<1024, blk, 0, stream>>>(x_emb, t_emb, comb);
  // per-feature processor collapses to per-batch (comb independent of f)
  gemm(comb, fp_w1, fp_b1, h1, 512, 512, 1024, 1);
  gemm(h1, fp_w2, fp_b2, h2, 512, 512, 512, 1);
  gemm(h2, fp_w3, fp_b3, h3, 512, 512, 512, 0);
  // rank-structured q/k/v: q[b,f] = (h3[b]@wq+bq) + (0.1*fid_pad[f]@wq)
  make_fs<<<256, blk, 0, stream>>>(fid, fs);
  gemm(h3, wq, bq, qh, 512, 512, 512, 0);
  gemm(h3, wk, bk, khb, 512, 512, 512, 0);
  gemm(h3, wv, bvp, vhb, 512, 512, 512, 0);
  gemm(fs, wq, nullptr, qf, 128, 512, 512, 0);
  gemm(fs, wk, nullptr, kfb, 128, 512, 512, 0);
  gemm(fs, wv, nullptr, vfb, 128, 512, 512, 0);
  // sparse attention + feature-mean pooling (mean commutes with wo projection)
  attn_kernel<<<dim3(8, 512), blk, 0, stream>>>(qh, qf, khb, kfb, vhb, vfb, pooled);
  gemm(pooled, wo, bo, g0, 512, 512, 512, 0);
  // global processor + LayerNorm
  gemm(g0, gp_w1, gp_b1, gp1, 512, 1024, 512, 1);
  gemm(gp1, gp_w2, gp_b2, g2, 512, 512, 1024, 0);
  ln_kernel<<<512, blk, 0, stream>>>(g2, ln_g, ln_b, gn);
  // denoise net -> tanh head
  gemm(gn, dn_w1, dn_b1, r1, 512, 1024, 512, 1);
  gemm(r1, dn_w2, dn_b2, r2, 512, 1024, 1024, 1);
  gemm(r2, dn_w3, dn_b3, r3, 512, 256, 1024, 0);
  tanh_head<<<256, blk, 0, stream>>>(r3, (float*)d_out);
}
